// BertSelfAttention_72164040507878
// MI455X (gfx1250) — matrix operand order repaired and checked by reading, not compiled, not run on hardware
//
#include <hip/hip_runtime.h>

// ---------------------------------------------------------------------------
// BERT self-attention for MI455X (gfx1250, wave32, WMMA bf16).
// B=8, D=1024, L=1024, H=16, DH=64.  Bandwidth-bound => bf16 intermediates,
// v_wmma_f32_16x16x32_bf16 with fp32 accumulation, LDS-resident score slab.
// ---------------------------------------------------------------------------

typedef __attribute__((ext_vector_type(16))) __bf16        v16bf;
typedef __attribute__((ext_vector_type(8)))  float         v8f;
typedef __attribute__((ext_vector_type(4)))  unsigned int  v4u;

constexpr int Bn  = 8;
constexpr int Dn  = 1024;
constexpr int Ln  = 1024;
constexpr int Hn  = 16;
constexpr int DHn = 64;

union Frag {
    v4u          q[2];
    unsigned int u[8];
    v16bf        v;
};

__device__ __forceinline__ unsigned short f2bf(float f) {
    unsigned int u = __float_as_uint(f);
    u += 0x7FFFu + ((u >> 16) & 1u);        // round-to-nearest-even
    return (unsigned short)(u >> 16);
}
__device__ __forceinline__ unsigned int pack2bf(float a, float b) {
    return (unsigned int)f2bf(a) | ((unsigned int)f2bf(b) << 16);
}

__device__ __forceinline__ v8f wmma_bf16(v16bf a, v16bf b, v8f c) {
    // D = A(16x32) * B(32x16) + C, fp32 accumulate
    return __builtin_amdgcn_wmma_f32_16x16x32_bf16(
        false, a, false, b, (short)0, c, false, false);
}

// A-matrix fragment (16x32 bf16).  rowptr_k0 = &row[m][k0], row K-contiguous.
// VGPR i holds K = (i>>2)*16 + khalf*8 + (i&3)*2 (pairs).
__device__ __forceinline__ v16bf load_frag_a(const unsigned short* rowptr_k0, int khalf) {
    Frag f;
    f.q[0] = *(const v4u*)(rowptr_k0 + khalf * 8);
    f.q[1] = *(const v4u*)(rowptr_k0 + 16 + khalf * 8);
    return f.v;
}
// B-matrix fragment (32x16 bf16).  rowptr_k0 = &col-n storage row (K-contiguous).
// VGPR i holds K = (i>>2)*16 + khalf*8 + (i&3)*2 (pairs).
__device__ __forceinline__ v16bf load_frag_b(const unsigned short* rowptr_k0, int khalf) {
    Frag f;
    const unsigned short* p = rowptr_k0 + khalf * 8;
    f.q[0] = *(const v4u*)(p);
    f.q[1] = *(const v4u*)(p + 16);
    return f.v;
}

// ---------------------------------------------------------------------------
// Kernel 1: fp32 -> bf16 weight pack.  Wc = [3][D][D] (q,k,v), row-major.
// ---------------------------------------------------------------------------
__global__ __launch_bounds__(256) void cvt_weights(
    const float* __restrict__ Wq, const float* __restrict__ Wk,
    const float* __restrict__ Wv, unsigned short* __restrict__ Wc)
{
    size_t i = ((size_t)blockIdx.x * 256 + threadIdx.x) * 2;   // even index
    size_t n = (size_t)3 * Dn * Dn;
    if (i >= n) return;
    size_t mat = i / ((size_t)Dn * Dn);
    size_t off = i - mat * (size_t)Dn * Dn;
    const float* W = (mat == 0) ? Wq : (mat == 1 ? Wk : Wv);
    *(unsigned int*)(Wc + i) = pack2bf(W[off], W[off + 1]);
}

// ---------------------------------------------------------------------------
// Kernel 2: hidden_states [B,D,L] fp32 -> hsT [B,L,D] bf16 (transposed so the
// projection GEMM B-matrix is K-contiguous).
// ---------------------------------------------------------------------------
__global__ __launch_bounds__(256) void transpose_hs(
    const float* __restrict__ hs, unsigned short* __restrict__ hsT)
{
    size_t i = (size_t)blockIdx.x * 256 + threadIdx.x;   // over B*D*L
    int l = (int)(i % Ln);
    size_t bd = i / Ln;
    int d = (int)(bd % Dn);
    int b = (int)(bd / Dn);
    hsT[((size_t)b * Ln + l) * Dn + d] = f2bf(hs[i]);
}

// ---------------------------------------------------------------------------
// Kernel 3: fused QKV projection GEMM.  One wave computes a 64(o) x 32(l) tile.
//   Y[b,o,l] = sum_d W[o,d] * hs[b,d,l] + bias[o]
// A = W (row-major, K-contig), B = hsT[b] (row n = position l, K-contig).
// Q,K stored [B,H,L,DH] bf16;  V stored [B,D,L] bf16 (natural).
// grid = (L/32, D/64, 3*B), block = 32.
// ---------------------------------------------------------------------------
__global__ __launch_bounds__(32) void qkv_gemm(
    const unsigned short* __restrict__ hsT, const unsigned short* __restrict__ Wc,
    const float* __restrict__ bq, const float* __restrict__ bk,
    const float* __restrict__ bv,
    unsigned short* __restrict__ Qo, unsigned short* __restrict__ Ko,
    unsigned short* __restrict__ Vo)
{
    const int lane  = threadIdx.x;
    const int m15   = lane & 15;
    const int khalf = lane >> 4;

    const int l0  = blockIdx.x * 32;
    const int o0  = blockIdx.y * 64;
    const int zb  = blockIdx.z;
    const int b   = zb / 3;
    const int mat = zb - b * 3;

    const unsigned short* A  = Wc + (size_t)mat * Dn * Dn;
    const unsigned short* Bm = hsT + (size_t)b * Ln * Dn;
    const float* bias = (mat == 0) ? bq : (mat == 1 ? bk : bv);

    const unsigned short* Arow[4];
#pragma unroll
    for (int mi = 0; mi < 4; ++mi)
        Arow[mi] = A + (size_t)(o0 + mi * 16 + m15) * Dn;
    const unsigned short* Brow[2];
#pragma unroll
    for (int ni = 0; ni < 2; ++ni)
        Brow[ni] = Bm + (size_t)(l0 + ni * 16 + m15) * Dn;

    v8f acc[4][2];
#pragma unroll
    for (int mi = 0; mi < 4; ++mi)
#pragma unroll
        for (int ni = 0; ni < 2; ++ni)
            acc[mi][ni] = (v8f){};

    for (int k0 = 0; k0 < Dn; k0 += 32) {
        if (k0 + 64 < Dn) {
            __builtin_prefetch(Arow[0] + k0 + 64, 0, 1);   // global_prefetch_b8
            __builtin_prefetch(Brow[0] + k0 + 64, 0, 1);
        }
        v16bf af[4], bf[2];
#pragma unroll
        for (int mi = 0; mi < 4; ++mi) af[mi] = load_frag_a(Arow[mi] + k0, khalf);
#pragma unroll
        for (int ni = 0; ni < 2; ++ni) bf[ni] = load_frag_b(Brow[ni] + k0, khalf);
#pragma unroll
        for (int mi = 0; mi < 4; ++mi)
#pragma unroll
            for (int ni = 0; ni < 2; ++ni)
                acc[mi][ni] = wmma_bf16(af[mi], bf[ni], acc[mi][ni]);
    }

    // Writeback: C/D layout: N = lane&15, M = i + 8*(lane>>4)
#pragma unroll
    for (int mi = 0; mi < 4; ++mi) {
#pragma unroll
        for (int ni = 0; ni < 2; ++ni) {
#pragma unroll
            for (int i = 0; i < 8; ++i) {
                int oo = o0 + mi * 16 + i + 8 * khalf;
                int ll = l0 + ni * 16 + m15;
                float y = acc[mi][ni][i] + bias[oo];
                unsigned short v = f2bf(y);
                if (mat == 2) {
                    Vo[(size_t)(b * Dn + oo) * Ln + ll] = v;     // [B,D,L]
                } else {
                    int h = oo >> 6, dh = oo & 63;
                    unsigned short* dst = (mat == 0) ? Qo : Ko;  // [B,H,L,DH]
                    dst[(((size_t)(b * Hn + h) * Ln) + ll) * DHn + dh] = v;
                }
            }
        }
    }
}

// ---------------------------------------------------------------------------
// Kernel 4: fused attention for one (b, h, 16-query block).
//   scores = Q K^T / 8 + mask  -> LDS fp32 slab [16][1024]
//   softmax in LDS             -> bf16 prob slab [16][512] (packed pairs)
//   ctx = P V (fp32 accum)     -> out[b, h*64+dh, q]
// block = 128 (4 waves), grid = (L/16, H, B).
// ---------------------------------------------------------------------------
__global__ __launch_bounds__(128) void attention(
    const unsigned short* __restrict__ Q, const unsigned short* __restrict__ K,
    const unsigned short* __restrict__ V, const float* __restrict__ mask,
    float* __restrict__ out)
{
    extern __shared__ float smem[];
    float*        S       = smem;                            // 16*1024 fp32
    unsigned int* P       = (unsigned int*)(smem + 16 * 1024); // 16*512 u32
    float*        red     = smem + 16 * 1024 + 16 * 512;     // 128
    float*        rowmax  = red + 128;                       // 16
    float*        rowrinv = rowmax + 16;                     // 16

    const int t     = threadIdx.x;
    const int lane  = t & 31;
    const int wv    = t >> 5;         // wave 0..3
    const int m15   = lane & 15;
    const int khalf = lane >> 4;

    const int q0 = blockIdx.x * 16;
    const int h  = blockIdx.y;
    const int b  = blockIdx.z;

    const unsigned short* Qb = Q + ((size_t)(b * Hn + h) * Ln) * DHn;
    const unsigned short* Kb = K + ((size_t)(b * Hn + h) * Ln) * DHn;
    const float*       maskb = mask + (size_t)b * Ln * Ln;   // [B,1,L,L]

    // Q fragments (A-matrix, M = query row, Kdim = DH): two 16x32 chunks
    v16bf qa0 = load_frag_a(Qb + (size_t)(q0 + m15) * DHn + 0,  khalf);
    v16bf qa1 = load_frag_a(Qb + (size_t)(q0 + m15) * DHn + 32, khalf);

    const float scale = 0.125f;       // 1/sqrt(DH)

    // ---- scores: each wave owns key tiles n0 = wv*16 + 64*j ----
    for (int n0 = wv * 16; n0 < Ln; n0 += 64) {
        v16bf kb0 = load_frag_b(Kb + (size_t)(n0 + m15) * DHn + 0,  khalf);
        v16bf kb1 = load_frag_b(Kb + (size_t)(n0 + m15) * DHn + 32, khalf);
        v8f sacc = (v8f){};
        sacc = wmma_bf16(qa0, kb0, sacc);
        sacc = wmma_bf16(qa1, kb1, sacc);
        int nn = n0 + m15;
#pragma unroll
        for (int i = 0; i < 8; ++i) {
            int mrow = i + 8 * khalf;
            float sv = sacc[i] * scale + maskb[(size_t)(q0 + mrow) * Ln + nn];
            S[mrow * Ln + nn] = sv;
        }
    }
    __syncthreads();

    // ---- softmax stats (16 rows x 1024, 8 threads per row) ----
    const int r     = t >> 3;
    const int chunk = t & 7;
    const float* Srow = S + r * Ln + chunk * 128;
    float mx = -3.0e38f;
    for (int i = 0; i < 128; ++i) mx = fmaxf(mx, Srow[i]);
    red[t] = mx;
    __syncthreads();
    if (t < 16) {
        float m = red[t * 8];
        for (int j = 1; j < 8; ++j) m = fmaxf(m, red[t * 8 + j]);
        rowmax[t] = m;
    }
    __syncthreads();
    float rm = rowmax[r];
    float sum = 0.f;
    for (int i = 0; i < 128; ++i) sum += __expf(Srow[i] - rm);
    red[t] = sum;
    __syncthreads();
    if (t < 16) {
        float s = red[t * 8];
        for (int j = 1; j < 8; ++j) s += red[t * 8 + j];
        rowrinv[t] = 1.0f / s;
    }
    __syncthreads();

    // ---- build bf16 probability slab P[16][512] (packed key pairs) ----
    for (int idx = t; idx < 16 * 512; idx += 128) {
        int rr = idx >> 9, cc = idx & 511;
        float rmx = rowmax[rr], ri = rowrinv[rr];
        float p0 = __expf(S[rr * Ln + 2 * cc]     - rmx) * ri;
        float p1 = __expf(S[rr * Ln + 2 * cc + 1] - rmx) * ri;
        P[idx] = pack2bf(p0, p1);
    }
    __syncthreads();

    // ---- ctx = P(16 x L) * V(L x DH); wave wv owns dh tile [wv*16, wv*16+16) ----
    const unsigned short* Vrow =
        V + ((size_t)(b * Dn + h * DHn + wv * 16 + m15)) * Ln;  // V[b, h*64+dh, :]
    const unsigned int* Prow = P + m15 * 512;                   // A row m (packed)
    v8f cacc = (v8f){};
    for (int k0 = 0; k0 < Ln; k0 += 32) {
        Frag fa;                                   // A frag from LDS probs
        int base = (k0 >> 1) + khalf * 4;          // u32 index: K/2
        fa.q[0] = *(const v4u*)(Prow + base);      // ds_load_b128
        fa.q[1] = *(const v4u*)(Prow + base + 8);
        v16bf vb = load_frag_b(Vrow + k0, khalf);  // key-contiguous
        cacc = wmma_bf16(fa.v, vb, cacc);
    }

    // out[b, h*64+dh, q]: N = dh (lane), M = query offset
    int dh = wv * 16 + m15;
    float* orow = out + ((size_t)(b * Dn + h * DHn + dh)) * Ln + q0;
#pragma unroll
    for (int i = 0; i < 8; ++i) orow[i + 8 * khalf] = cacc[i];
}

// ---------------------------------------------------------------------------
// Launch.  Workspace (bf16):
//   hsT [B,L,D] 16MB | Wc [3,D,D] 6MB | Q [B,H,L,DH] 16MB | K 16MB | V [B,D,L] 16MB
// ---------------------------------------------------------------------------
extern "C" void kernel_launch(void* const* d_in, const int* in_sizes, int n_in,
                              void* d_out, int out_size, void* d_ws, size_t ws_size,
                              hipStream_t stream) {
    const float* hs   = (const float*)d_in[0];
    const float* mask = (const float*)d_in[1];
    const float* Wq   = (const float*)d_in[2];
    const float* bq   = (const float*)d_in[3];
    const float* Wk   = (const float*)d_in[4];
    const float* bk   = (const float*)d_in[5];
    const float* Wv   = (const float*)d_in[6];
    const float* bv   = (const float*)d_in[7];
    float* out = (float*)d_out;

    unsigned short* hsT = (unsigned short*)d_ws;
    unsigned short* Wc  = hsT + (size_t)Bn * Ln * Dn;        // +8M
    unsigned short* Qb  = Wc  + (size_t)3 * Dn * Dn;         // +3M
    unsigned short* Kb  = Qb  + (size_t)Bn * Hn * Ln * DHn;  // +8M
    unsigned short* Vb  = Kb  + (size_t)Bn * Hn * Ln * DHn;  // +8M

    // 1. weights fp32 -> bf16
    cvt_weights<<<(3 * Dn * Dn / 2 + 255) / 256, 256, 0, stream>>>(Wq, Wk, Wv, Wc);
    // 2. hidden fp32 [B,D,L] -> bf16 [B,L,D]
    transpose_hs<<<(Bn * Dn * Ln) / 256, 256, 0, stream>>>(hs, hsT);
    // 3. fused QKV projection GEMM
    qkv_gemm<<<dim3(Ln / 32, Dn / 64, 3 * Bn), 32, 0, stream>>>(
        hsT, Wc, bq, bk, bv, Qb, Kb, Vb);
    // 4. fused attention (scores + softmax + PV), 97 KB dynamic LDS
    size_t smem = (size_t)(16 * 1024 + 16 * 512 + 128 + 16 + 16) * sizeof(float);
    attention<<<dim3(Ln / 16, Hn, Bn), 128, smem, stream>>>(Qb, Kb, Vb, mask, out);
}
